// DSHybridBitMLP_3118146256932
// MI455X (gfx1250) — 8-bit matrix operand order repaired and checked by reading, compiled, not run on hardware
//
#include <hip/hip_runtime.h>
#include <hip/hip_bf16.h>
#include <stdint.h>


typedef int v2i __attribute__((ext_vector_type(2)));
typedef int v4i __attribute__((ext_vector_type(4)));
typedef int v8i __attribute__((ext_vector_type(8)));

#define HIDDEN 2048
#define INTER  5632
#define NTOK   (4 * 2048)
#define NGATE  (2 * INTER)
#define EPSV   1e-5f

__global__ __launch_bounds__(256)
void abssum_partial_kernel(const float* __restrict__ w, size_t n,
                           float* __restrict__ partial) {
    float s = 0.f;
    size_t stride = (size_t)gridDim.x * blockDim.x;
    for (size_t i = (size_t)blockIdx.x * blockDim.x + threadIdx.x; i < n; i += stride)
        s += fabsf(w[i]);
    __shared__ float sm[256];
    sm[threadIdx.x] = s;
    __syncthreads();
    for (int t = 128; t > 0; t >>= 1) {
        if ((int)threadIdx.x < t) sm[threadIdx.x] += sm[threadIdx.x + t];
        __syncthreads();
    }
    if (threadIdx.x == 0) partial[blockIdx.x] = sm[0];
}

__global__ __launch_bounds__(256)
void absmean_final_kernel(const float* __restrict__ partial, int nparts,
                          float inv_count, float* __restrict__ sc) {
    float s = 0.f;
    for (int i = threadIdx.x; i < nparts; i += blockDim.x) s += partial[i];
    __shared__ float sm[256];
    sm[threadIdx.x] = s;
    __syncthreads();
    for (int t = 128; t > 0; t >>= 1) {
        if ((int)threadIdx.x < t) sm[threadIdx.x] += sm[threadIdx.x + t];
        __syncthreads();
    }
    if (threadIdx.x == 0) {
        float mean = fmaxf(sm[0] * inv_count, 1e-5f);
        sc[0] = 1.f / mean;
        sc[1] = mean;
    }
}

__global__ __launch_bounds__(256)
void quant_weight_kernel(const float* __restrict__ w, size_t n,
                         const float* __restrict__ sc,
                         int8_t* __restrict__ q) {
    float scale = sc[0];
    size_t stride = (size_t)gridDim.x * blockDim.x;
    for (size_t i = (size_t)blockIdx.x * blockDim.x + threadIdx.x; i < n; i += stride) {
        float v = fminf(fmaxf(rintf(w[i] * scale), -1.f), 1.f);
        q[i] = (int8_t)(int)v;
    }
}

__global__ __launch_bounds__(256)
void rmsnorm_quant_kernel(const float* __restrict__ x,
                          const float* __restrict__ g,
                          int8_t* __restrict__ q,
                          float* __restrict__ descale,
                          int K) {
    int row = blockIdx.x;
    const float* xr = x + (size_t)row * K;
    float ss = 0.f, amax = 0.f;
    for (int i = threadIdx.x; i < K; i += blockDim.x) {
        float v = xr[i];
        ss += v * v;
        amax = fmaxf(amax, fabsf(v * g[i]));
    }
    __shared__ float s_ss[256], s_mx[256];
    s_ss[threadIdx.x] = ss;
    s_mx[threadIdx.x] = amax;
    __syncthreads();
    for (int t = 128; t > 0; t >>= 1) {
        if ((int)threadIdx.x < t) {
            s_ss[threadIdx.x] += s_ss[threadIdx.x + t];
            s_mx[threadIdx.x] = fmaxf(s_mx[threadIdx.x], s_mx[threadIdx.x + t]);
        }
        __syncthreads();
    }
    float rstd = rsqrtf(s_ss[0] / (float)K + EPSV);
    float mx = fmaxf(s_mx[0] * rstd, 1e-5f);
    float scale = 127.f / mx;
    if (threadIdx.x == 0) descale[row] = mx * (1.f / 127.f);
    int8_t* qr = q + (size_t)row * K;
    for (int i = threadIdx.x; i < K; i += blockDim.x) {
        float vn = xr[i] * rstd * g[i];
        float qq = fminf(fmaxf(rintf(vn * scale), -128.f), 127.f);
        qr[i] = (int8_t)(int)qq;
    }
}

__device__ __forceinline__ v8i load_a_frag(const int8_t* p) {
    v4i a0 = *(const v4i*)(p);
    v4i a1 = *(const v4i*)(p + 32);
    v8i f = {a0.x, a0.y, a0.z, a0.w, a1.x, a1.y, a1.z, a1.w};
    return f;
}

__device__ __forceinline__ v8i load_b_frag(const int8_t* p) {
    v4i b0 = *(const v4i*)(p);
    v4i b1 = *(const v4i*)(p + 32);
    v8i f = {b0.x, b0.y, b0.z, b0.w, b1.x, b1.y, b1.z, b1.w};
    return f;
}

template <bool SWIGLU>
__global__ __launch_bounds__(256)
void bitgemm_iu8_kernel(const int8_t* __restrict__ aq,
                        const int8_t* __restrict__ wq,
                        const float* __restrict__ a_ds,
                        const float* __restrict__ w_sc,
                        float* __restrict__ out,
                        int K, int Nout) {
    const int lane = threadIdx.x & 31;
    const int wave = threadIdx.x >> 5;
    const int nl   = lane & 15;
    const int half = lane >> 4;

    const int m0 = blockIdx.y * 64;
    int n0, n1;
    if (SWIGLU) {
        n0 = (blockIdx.x * 8 + wave) * 16;
        n1 = n0 + Nout;
    } else {
        n0 = blockIdx.x * 256 + wave * 16;
        n1 = n0 + 128;
    }

    const size_t mstride = (size_t)16 * K;
    const int8_t* aptr  = aq + (size_t)(m0 + nl) * K + half * 16;
    const int8_t* bptr0 = wq + (size_t)(n0 + nl) * K + half * 16;
    const int8_t* bptr1 = wq + (size_t)(n1 + nl) * K + half * 16;

    v8i acc0[4], acc1[4];
#pragma unroll
    for (int mt = 0; mt < 4; ++mt) {
        acc0[mt] = (v8i){0, 0, 0, 0, 0, 0, 0, 0};
        acc1[mt] = (v8i){0, 0, 0, 0, 0, 0, 0, 0};
    }

    for (int k = 0; k < K; k += 64) {
        v8i b0 = load_b_frag(bptr0 + k);
        v8i b1 = load_b_frag(bptr1 + k);
#pragma unroll
        for (int mt = 0; mt < 4; ++mt) {
            v8i a = load_a_frag(aptr + mt * mstride + k);
            acc0[mt] = __builtin_amdgcn_wmma_i32_16x16x64_iu8(
                 true, a,  true, b0, acc0[mt],
                 false,  false);
            acc1[mt] = __builtin_amdgcn_wmma_i32_16x16x64_iu8(
                true, a, true, b1, acc1[mt], false, false);
        }
    }

    const float wds = w_sc[1];
#pragma unroll
    for (int mt = 0; mt < 4; ++mt) {
#pragma unroll
        for (int r = 0; r < 8; ++r) {
            int m = m0 + mt * 16 + r + half * 8;
            float ds = a_ds[m] * wds;
            float v0 = (float)acc0[mt][r] * ds;
            float v1 = (float)acc1[mt][r] * ds;
            if (SWIGLU) {
                float hv = v0 * (1.f / (1.f + expf(-v0))) * v1;
                out[(size_t)m * Nout + (n0 + nl)] = hv;
            } else {
                out[(size_t)m * Nout + (n0 + nl)] = v0;
                out[(size_t)m * Nout + (n1 + nl)] = v1;
            }
        }
    }
}

extern "C" void kernel_launch(void* const* d_in, const int* in_sizes, int n_in,
                              void* d_out, int out_size, void* d_ws, size_t ws_size,
                              hipStream_t stream) {
    (void)in_sizes; (void)n_in; (void)out_size; (void)ws_size;

    const float* x    = (const float*)d_in[0];
    const float* gw   = (const float*)d_in[1];
    const float* dw   = (const float*)d_in[2];
    const float* gnw  = (const float*)d_in[3];
    const float* dnw  = (const float*)d_in[4];
    float* out = (float*)d_out;

    size_t off = 0;
    auto take = [&](size_t bytes) -> char* {
        char* p = (char*)d_ws + off;
        off += (bytes + 255) & ~(size_t)255;
        return p;
    };
    int8_t* qgw  = (int8_t*)take((size_t)NGATE * HIDDEN);
    int8_t* qdw  = (int8_t*)take((size_t)HIDDEN * INTER);
    int8_t* a1q  = (int8_t*)take((size_t)NTOK * HIDDEN);
    int8_t* a2q  = (int8_t*)take((size_t)NTOK * INTER);
    float*  a1ds = (float*)take((size_t)NTOK * sizeof(float));
    float*  a2ds = (float*)take((size_t)NTOK * sizeof(float));
    float*  h    = (float*)take((size_t)NTOK * INTER * sizeof(float));
    float*  p1   = (float*)take(1024 * sizeof(float));
    float*  p2   = (float*)take(1024 * sizeof(float));
    float*  gsc  = (float*)take(2 * sizeof(float));
    float*  dsc  = (float*)take(2 * sizeof(float));

    const size_t ngw = (size_t)NGATE * HIDDEN;
    const size_t ndw = (size_t)HIDDEN * INTER;

    abssum_partial_kernel<<<1024, 256, 0, stream>>>(gw, ngw, p1);
    absmean_final_kernel<<<1, 256, 0, stream>>>(p1, 1024, 1.f / (float)ngw, gsc);
    quant_weight_kernel<<<2048, 256, 0, stream>>>(gw, ngw, gsc, qgw);

    abssum_partial_kernel<<<1024, 256, 0, stream>>>(dw, ndw, p2);
    absmean_final_kernel<<<1, 256, 0, stream>>>(p2, 1024, 1.f / (float)ndw, dsc);
    quant_weight_kernel<<<2048, 256, 0, stream>>>(dw, ndw, dsc, qdw);

    rmsnorm_quant_kernel<<<NTOK, 256, 0, stream>>>(x, gnw, a1q, a1ds, HIDDEN);
    bitgemm_iu8_kernel<true><<<dim3(INTER / 16 / 8, NTOK / 64), 256, 0, stream>>>(
        a1q, qgw, a1ds, gsc, h, HIDDEN, INTER);

    rmsnorm_quant_kernel<<<NTOK, 256, 0, stream>>>(h, dnw, a2q, a2ds, INTER);
    bitgemm_iu8_kernel<false><<<dim3(HIDDEN / 256, NTOK / 64), 256, 0, stream>>>(
        a2q, qdw, a2ds, dsc, out, INTER, HIDDEN);
}
